// MultiHeadedAttention_33578054320510
// MI455X (gfx1250) — compile-verified, matrix operand order checked by reading, not run on hardware
//
#include <hip/hip_runtime.h>
#include <cstdint>

// ---------------------------------------------------------------------------
// MultiHeadedAttention for MI455X (gfx1250): bf16 WMMA + TDM async staging.
//   B=4, S=2048, D=1024, H=16, d_k=64.  fp32 in/out, bf16 internal.
// ---------------------------------------------------------------------------

typedef __attribute__((ext_vector_type(16))) __bf16 v16bf;
typedef __attribute__((ext_vector_type(8)))  float  v8f;

#define D_MODEL 1024
#define SEQ     2048
#define BATCH   4
#define HEADS   16
#define DK      64
#define SCLD    (SEQ + 4)   // padded score row stride (bank-conflict-free)
#define KVLD    66          // TDM-padded K/V tile row stride (64 + 1 DWORD)
#define WLD     34          // TDM-padded W tile row stride (32 + 1 DWORD)

// Device-pass-only probe: host clang has no AMDGPU builtin table.
#if defined(__HIP_DEVICE_COMPILE__)
#if !__has_builtin(__builtin_amdgcn_tensor_load_to_lds)
#error "gfx1250 device pass lacks __builtin_amdgcn_tensor_load_to_lds"
#endif
#endif

#if defined(__HIP_DEVICE_COMPILE__)
#if __has_builtin(__builtin_amdgcn_s_wait_tensorcnt)
#define TENSOR_WAIT(n) __builtin_amdgcn_s_wait_tensorcnt(n)
#else
#define TENSOR_WAIT(n) asm volatile("s_wait_tensorcnt " #n ::: "memory")
#endif
#else
#define TENSOR_WAIT(n) ((void)0)
#endif
#define COMPILER_FENCE() asm volatile("" ::: "memory")

__device__ __forceinline__ __bf16 to_bf16(float x)  { return (__bf16)x; }
__device__ __forceinline__ __bf16 to_bf16(__bf16 x) { return x; }

// K-offset of the element pair held in VGPR slot v (0..7) for lane-half hi,
// per CDNA5 ISA 7.12.2 (16-bit A-matrix 16x32 layout).
__device__ __forceinline__ int kslot(int v, int hi) {
  return ((v & 4) << 2) + ((v & 3) << 1) + (hi << 3);
}

// A-style fragment (also B when the B column is a contiguous row in memory).
template <typename T>
__device__ __forceinline__ v16bf load_frag_rows(const T* __restrict__ base,
                                                int ld, int lane) {
  const int m = lane & 15, hi = lane >> 4;
  const T* p = base + (size_t)m * ld;
  v16bf f;
#pragma unroll
  for (int v = 0; v < 8; ++v) {
    const int k = kslot(v, hi);
    f[2 * v]     = to_bf16(p[k]);
    f[2 * v + 1] = to_bf16(p[k + 1]);
  }
  return f;
}

// B fragment from a row-major [K][N] tile (e.g. V): lane holds column lane&15.
template <typename T>
__device__ __forceinline__ v16bf load_frag_cols(const T* __restrict__ base,
                                                int ld, int lane) {
  const int n = lane & 15, hi = lane >> 4;
  v16bf f;
#pragma unroll
  for (int v = 0; v < 8; ++v) {
    const int k = kslot(v, hi);
    f[2 * v]     = to_bf16(base[(size_t)k * ld + n]);
    f[2 * v + 1] = to_bf16(base[(size_t)(k + 1) * ld + n]);
  }
  return f;
}

__device__ __forceinline__ v8f wmma_bf16(v16bf a, v16bf b, v8f c) {
  return __builtin_amdgcn_wmma_f32_16x16x32_bf16(false, a, false, b,
                                                 (short)0, c, false, false);
}

// ---------------------------------------------------------------------------
// Tensor Data Mover: async 2-D bf16 tile load (rows x cols, row stride in
// elements) global -> LDS, with per-row LDS padding (bank spread).
// D# bitfields per ISA ch.8.  pad_icode: interval = 2^(code+1) DWORDs;
// pad_amount fixed at 1 DWORD.
// ---------------------------------------------------------------------------
typedef __attribute__((ext_vector_type(4))) unsigned int u32x4;
typedef __attribute__((ext_vector_type(4))) int          i32x4;
typedef __attribute__((ext_vector_type(8))) int          i32x8;

__device__ __forceinline__ int rfl(unsigned int x) {
#if defined(__HIP_DEVICE_COMPILE__)
  return __builtin_amdgcn_readfirstlane((int)x);
#else
  return (int)x;
#endif
}

__device__ __forceinline__ unsigned int lds_off(const void* p) {
  // generic pointer to LDS: addr[31:0] is the LDS byte address (flat mapping)
  return (unsigned int)(uintptr_t)p;
}

__device__ __forceinline__ void tdm_load_tile_bf16(unsigned int lds_byte,
                                                   const void* gptr, int cols,
                                                   int rows, int stride_elems,
                                                   int pad_icode) {
  const unsigned long long ga = (unsigned long long)(uintptr_t)gptr;
  u32x4 g0;
  g0[0] = (unsigned)rfl(1u);                 // count=1 (valid user descriptor)
  g0[1] = (unsigned)rfl(lds_byte);           // lds_addr
  g0[2] = (unsigned)rfl((unsigned)(ga & 0xffffffffu));         // gaddr[95:64]
  g0[3] = (unsigned)rfl((unsigned)(((ga >> 32) & 0x01ffffffu)  // gaddr[120:96]
                                   | (2u << 30)));             // type=2 image
  const unsigned td = 1u << 20;              // generous tensor dims (no OOB)
  const unsigned padw =
      (1u << 20) | ((unsigned)pad_icode << 22);  // pad_enable | interval, amt=1
  i32x8 g1;
  g1[0] = rfl((1u << 16) | padw);            // data_size=2B | LDS padding
  g1[1] = rfl((td & 0xffffu) << 16);         // tensor_dim0[15:0]
  g1[2] = rfl((td >> 16) | ((td & 0xffffu) << 16));   // dim0 hi | dim1 lo
  g1[3] = rfl((td >> 16) | ((unsigned)cols << 16));   // dim1 hi | tile_dim0
  g1[4] = rfl((unsigned)rows);               // tile_dim1 (tile_dim2 = 0)
  g1[5] = rfl((unsigned)stride_elems);       // tensor_dim0_stride[31:0]
  g1[6] = 0;
  g1[7] = 0;
  i32x4 z4 = {0, 0, 0, 0};
#if defined(__HIP_DEVICE_COMPILE__)
#if defined(__clang_major__) && __clang_major__ >= 23
  i32x8 z8 = {0, 0, 0, 0, 0, 0, 0, 0};
  __builtin_amdgcn_tensor_load_to_lds(g0, g1, z4, z4, z8, 0);
#else
  __builtin_amdgcn_tensor_load_to_lds(g0, g1, z4, z4, 0);
#endif
#else
  (void)g0; (void)g1; (void)z4;   // host pass: no-op
#endif
}

// ---------------------------------------------------------------------------
// fp32 -> bf16 weight conversion (one-shot, 4 x 1M elements)
// ---------------------------------------------------------------------------
__global__ __launch_bounds__(256) void cvt_bf16_kernel(
    const float* __restrict__ in, __bf16* __restrict__ out, int n) {
  const int i = (blockIdx.x * 256 + threadIdx.x) * 4;
  if (i + 3 < n) {
    const float4 f = *(const float4*)(in + i);
    out[i]     = (__bf16)f.x;
    out[i + 1] = (__bf16)f.y;
    out[i + 2] = (__bf16)f.z;
    out[i + 3] = (__bf16)f.w;
  }
}

// ---------------------------------------------------------------------------
// GEMM with bias:  Y[M,N] = X[M,K] @ W[N,K]^T + bias[N]     (W is bf16)
// 4-wave WG computes 128x64; each wave a 32x64 strip.  The shared 64x32 W
// tile is staged by the TDM (wave 0 drives, double-buffered) so the DMA of
// k-step j+1 overlaps the 8 WMMAs/wave of step j.
// ---------------------------------------------------------------------------
template <typename IN_T, typename OUT_T>
__global__ __launch_bounds__(128) void gemm_bias_kernel(
    const IN_T* __restrict__ X, const __bf16* __restrict__ W,
    const float* __restrict__ bias, OUT_T* __restrict__ Y,
    int M, int N, int K) {
  __shared__ __bf16 wtile[2][64 * WLD];  // TDM-padded rows (32+2 elems)

  const int lane = threadIdx.x & 31;
  const int wave = threadIdx.x >> 5;
  const int m0 = blockIdx.y * 128 + wave * 32;
  const int n0 = blockIdx.x * 64;

  if (wave == 0) {
    tdm_load_tile_bf16(lds_off(&wtile[0][0]), W + (size_t)n0 * K, 32, 64, K, 3);
    TENSOR_WAIT(0);
  }
  __syncthreads();

  v8f acc[2][4] = {};
  for (int k0 = 0; k0 < K; k0 += 32) {
    const int j = k0 >> 5;
    if (wave == 0 && k0 + 32 < K)
      tdm_load_tile_bf16(lds_off(&wtile[(j + 1) & 1][0]),
                         W + (size_t)n0 * K + k0 + 32, 32, 64, K, 3);
    const v16bf a0 = load_frag_rows(X + (size_t)m0 * K + k0, K, lane);
    const v16bf a1 = load_frag_rows(X + (size_t)(m0 + 16) * K + k0, K, lane);
    const __bf16* wt = &wtile[j & 1][0];
#pragma unroll
    for (int t = 0; t < 4; ++t) {
      const v16bf b = load_frag_rows(wt + (size_t)(16 * t) * WLD, WLD, lane);
      acc[0][t] = wmma_bf16(a0, b, acc[0][t]);
      acc[1][t] = wmma_bf16(a1, b, acc[1][t]);
    }
    if (wave == 0) TENSOR_WAIT(0);
    __syncthreads();
  }

  const int n = lane & 15, hi = lane >> 4;
#pragma unroll
  for (int t = 0; t < 4; ++t) {
    const int col = n0 + 16 * t + n;
    const float bv = bias[col];
#pragma unroll
    for (int s = 0; s < 2; ++s)
#pragma unroll
      for (int i = 0; i < 8; ++i) {
        const int row = m0 + 16 * s + i + 8 * hi;
        Y[(size_t)row * N + col] = (OUT_T)(acc[s][t][i] + bv);
      }
  }
}

// ---------------------------------------------------------------------------
// Attention core: one WG = (batch b, head h, 16 query rows).
// Scores for the full 2048-key row live in LDS; K tiles are TDM-staged
// per-wave (private TENSORcnt, no barriers), V tiles TDM-staged WG-wide.
// ---------------------------------------------------------------------------
__global__ __launch_bounds__(128) void attention_kernel(
    const __bf16* __restrict__ Q, const __bf16* __restrict__ Km,
    const __bf16* __restrict__ V, __bf16* __restrict__ Xo) {
  __shared__ float  sc[16][SCLD];          // ~128 KB scores (padded rows)
  __shared__ __bf16 kbuf[4][2][16 * KVLD]; // per-wave double-buffered K tiles
  __shared__ __bf16 vbuf[2][32 * KVLD];    // WG double-buffered V tiles
  __shared__ float  red_m[16][8];
  __shared__ float  red_s[16][8];

  const int lane = threadIdx.x & 31;
  const int wave = threadIdx.x >> 5;
  const int q0 = blockIdx.x * 16;
  const int h  = blockIdx.y;
  const int b  = blockIdx.z;
  const size_t rowbase = (size_t)b * SEQ;
  const int    hcol    = h * DK;

  // ---- scores: S[16 x 2048] = (Q tile) @ K^T * 1/sqrt(dk) -----------------
  const __bf16* Qb = Q + (rowbase + q0) * D_MODEL + hcol;
  const v16bf aq0 = load_frag_rows(Qb,      D_MODEL, lane);
  const v16bf aq1 = load_frag_rows(Qb + 32, D_MODEL, lane);

  const int n = lane & 15, hi = lane >> 4;
  const __bf16* Kbase = Km + rowbase * D_MODEL + hcol;
  const unsigned int kb[2] = {lds_off(&kbuf[wave][0][0]),
                              lds_off(&kbuf[wave][1][0])};
  const int NT = SEQ / 16 / 4;  // 32 key tiles per wave
  tdm_load_tile_bf16(kb[0], Kbase + (size_t)wave * 16 * D_MODEL, DK, 16,
                     D_MODEL, 4);
  for (int j = 0; j < NT; ++j) {
    const int t = wave + 4 * j;
    if (j + 1 < NT) {
      tdm_load_tile_bf16(kb[(j + 1) & 1],
                         Kbase + (size_t)(t + 4) * 16 * D_MODEL, DK, 16,
                         D_MODEL, 4);
      TENSOR_WAIT(1);     // current tile done, next one in flight
    } else {
      TENSOR_WAIT(0);
    }
    COMPILER_FENCE();
    const __bf16* kt = &kbuf[wave][j & 1][0];
    v16bf b0 = load_frag_rows(kt,      KVLD, lane);
    v16bf b1 = load_frag_rows(kt + 32, KVLD, lane);
    v8f c = {};
    c = wmma_bf16(aq0, b0, c);
    c = wmma_bf16(aq1, b1, c);
#pragma unroll
    for (int i = 0; i < 8; ++i)
      sc[i + 8 * hi][t * 16 + n] = c[i] * 0.125f;  // 1/sqrt(64)
  }
  __syncthreads();

  // ---- softmax over each 2048-wide row (8 threads per row) ----------------
  const int row = threadIdx.x >> 3;
  const int sub = threadIdx.x & 7;
  const int c0 = sub * (SEQ / 8), c1 = c0 + (SEQ / 8);

  float mx = -3.402823466e+38f;
  for (int c = c0; c < c1; ++c) mx = fmaxf(mx, sc[row][c]);
  red_m[row][sub] = mx;
  __syncthreads();

  float rmx = red_m[row][0];
#pragma unroll
  for (int i = 1; i < 8; ++i) rmx = fmaxf(rmx, red_m[row][i]);
  float sum = 0.f;
  for (int c = c0; c < c1; ++c) {
    float e = __expf(sc[row][c] - rmx);
    sc[row][c] = e;
    sum += e;
  }
  red_s[row][sub] = sum;
  __syncthreads();

  float rs = red_s[row][0];
#pragma unroll
  for (int i = 1; i < 8; ++i) rs += red_s[row][i];
  const float inv = 1.f / rs;
  for (int c = c0; c < c1; ++c) sc[row][c] *= inv;
  __syncthreads();

  // ---- O[16 x 64] = P @ V ; each wave owns a 16-wide d_k chunk ------------
  const int dk0 = wave * 16;
  const __bf16* Vbase = V + rowbase * D_MODEL + hcol;
  v8f acc = {};

  const unsigned int vb[2] = {lds_off(&vbuf[0][0]), lds_off(&vbuf[1][0])};
  if (wave == 0) {
    tdm_load_tile_bf16(vb[0], Vbase, DK, 32, D_MODEL, 4);
    TENSOR_WAIT(0);
  }
  __syncthreads();
  for (int j = 0; j < SEQ / 32; ++j) {
    if (wave == 0 && (j + 1) < SEQ / 32)
      tdm_load_tile_bf16(vb[(j + 1) & 1],
                         Vbase + (size_t)(j + 1) * 32 * D_MODEL, DK, 32,
                         D_MODEL, 4);
    // A = P[16][32j..32j+32) from LDS f32 -> bf16 (layout transpose via LDS)
    v16bf a;
    const int m = lane & 15;
#pragma unroll
    for (int v = 0; v < 8; ++v) {
      const int k = 32 * j + kslot(v, hi);
      a[2 * v]     = (__bf16)sc[m][k];
      a[2 * v + 1] = (__bf16)sc[m][k + 1];
    }
    const v16bf bb = load_frag_cols(&vbuf[j & 1][dk0], KVLD, lane);
    acc = wmma_bf16(a, bb, acc);
    if (wave == 0) TENSOR_WAIT(0);
    __syncthreads();
  }

#pragma unroll
  for (int i = 0; i < 8; ++i) {
    const size_t r = rowbase + q0 + i + 8 * hi;
    Xo[r * D_MODEL + hcol + dk0 + n] = (__bf16)acc[i];
  }
}

// ---------------------------------------------------------------------------
extern "C" void kernel_launch(void* const* d_in, const int* in_sizes, int n_in,
                              void* d_out, int out_size, void* d_ws,
                              size_t ws_size, hipStream_t stream) {
  const float* q  = (const float*)d_in[0];
  const float* k  = (const float*)d_in[1];
  const float* v  = (const float*)d_in[2];
  const float* Wq = (const float*)d_in[3];
  const float* bq = (const float*)d_in[4];
  const float* Wk = (const float*)d_in[5];
  const float* bk = (const float*)d_in[6];
  const float* Wv = (const float*)d_in[7];
  const float* bv = (const float*)d_in[8];
  const float* Wo = (const float*)d_in[9];
  const float* bo = (const float*)d_in[10];
  float* out = (float*)d_out;

  const int M = BATCH * SEQ;                 // 8192
  const size_t elems = (size_t)M * D_MODEL;  // 8M
  const int wn = D_MODEL * D_MODEL;          // 1M per weight

  __bf16* Qp    = (__bf16*)d_ws;             // 16 MB each
  __bf16* Kp    = Qp + elems;
  __bf16* Vp    = Kp + elems;
  __bf16* Xattn = Vp + elems;
  __bf16* Wqb   = Xattn + elems;             // 2 MB each
  __bf16* Wkb   = Wqb + wn;
  __bf16* Wvb   = Wkb + wn;
  __bf16* Wob   = Wvb + wn;

  const dim3 cvtGrid(wn / (256 * 4));
  cvt_bf16_kernel<<<cvtGrid, 256, 0, stream>>>(Wq, Wqb, wn);
  cvt_bf16_kernel<<<cvtGrid, 256, 0, stream>>>(Wk, Wkb, wn);
  cvt_bf16_kernel<<<cvtGrid, 256, 0, stream>>>(Wv, Wvb, wn);
  cvt_bf16_kernel<<<cvtGrid, 256, 0, stream>>>(Wo, Wob, wn);

  const dim3 gemmGrid(D_MODEL / 64, M / 128);  // (16, 64)
  const dim3 gemmBlk(128);
  gemm_bias_kernel<float, __bf16><<<gemmGrid, gemmBlk, 0, stream>>>(
      q, Wqb, bq, Qp, M, D_MODEL, D_MODEL);
  gemm_bias_kernel<float, __bf16><<<gemmGrid, gemmBlk, 0, stream>>>(
      k, Wkb, bk, Kp, M, D_MODEL, D_MODEL);
  gemm_bias_kernel<float, __bf16><<<gemmGrid, gemmBlk, 0, stream>>>(
      v, Wvb, bv, Vp, M, D_MODEL, D_MODEL);

  const dim3 attnGrid(SEQ / 16, HEADS, BATCH);  // (128, 16, 4)
  attention_kernel<<<attnGrid, gemmBlk, 0, stream>>>(Qp, Kp, Vp, Xattn);

  gemm_bias_kernel<__bf16, float><<<gemmGrid, gemmBlk, 0, stream>>>(
      Xattn, Wob, bo, out, M, D_MODEL, D_MODEL);
}
